// HybridPFNNRegressor_80762565034653
// MI455X (gfx1250) — matrix operand order repaired and checked by reading, not compiled, not run on hardware
//
#include <hip/hip_runtime.h>
#include <hip/hip_bf16.h>
#include <math.h>

typedef _Float16 v16h __attribute__((ext_vector_type(16)));
typedef _Float16 v8h  __attribute__((ext_vector_type(8)));
typedef float    v8f  __attribute__((ext_vector_type(8)));

// Workspace (f16) layout, in halves. All offsets are multiples of 16 halves
// (32 bytes) so every B-fragment load is 32B-aligned.
#define OFF_EXPW 0          // exp_w   : 128 x 32            =   4096
#define OFF_W1   4096       // w1 pad  : 256 x 192 (168->192)=  49152
#define OFF_PFW  53248      // pf_w    : 8 x 256 x 288       = 589824
#define OFF_WOUT 643072     // w_out   : 16 x 288            =   4608
#define WS_HALVES 647680    // ~1.3 MB total (L2-resident)

__device__ __forceinline__ float eluf(float x) {
  return x > 0.0f ? x : (__expf(x) - 1.0f);
}

__device__ __forceinline__ v8f wmma16(v16h a, v16h b, v8f c) {
  // D = A(16x32 f16) * B(32x16 f16) + C(16x16 f32)
  return __builtin_amdgcn_wmma_f32_16x16x32_f16(false, a, false, b,
                                                (short)0, c, false, false);
}

// A-fragment (16-bit A, 16x32), contiguous K order matching B: lanes 0-15 hold K=0..15;
// lanes 16-31 hold K=16..31 -> two adjacent 16B chunks.
__device__ __forceinline__ v16h load_a_frag(const _Float16* p) {
  v8h lo = *(const v8h*)(p);
  v8h hi = *(const v8h*)(p + 8);
  v16h r;
#pragma unroll
  for (int i = 0; i < 8; ++i) { r[i] = lo[i]; r[i + 8] = hi[i]; }
  return r;
}

// ---------------- one-shot weight conversion f32 -> f16 ----------------
__global__ void pfnn_convert_weights(const float* __restrict__ exp_w,
                                     const float* __restrict__ w1,
                                     const float* __restrict__ pf_w,
                                     const float* __restrict__ w_out,
                                     _Float16* __restrict__ ws) {
  const int i0 = blockIdx.x * blockDim.x + threadIdx.x;
  const int stride = gridDim.x * blockDim.x;
  for (int j = i0; j < 128 * 32; j += stride)
    ws[OFF_EXPW + j] = (_Float16)exp_w[j];
  for (int j = i0; j < 256 * 192; j += stride) {
    int n = j / 192, k = j % 192;
    ws[OFF_W1 + j] = (_Float16)(k < 168 ? w1[n * 168 + k] : 0.0f);
  }
  for (int j = i0; j < 8 * 256 * 288; j += stride)
    ws[OFF_PFW + j] = (_Float16)pf_w[j];
  for (int j = i0; j < 16 * 288; j += stride)
    ws[OFF_WOUT + j] = (_Float16)w_out[j];
}

// ---------------- fully fused PFNN forward, 32 rows / block ----------------
__global__ __launch_bounds__(256)
void pfnn_fused(const float* __restrict__ inputs,   // 16384 x 33
                const float* __restrict__ exp_b,    // 128
                const float* __restrict__ cp,       // 8 x 8 phase_node_cp
                const float* __restrict__ b1,       // 256
                const float* __restrict__ pf_b,     // 8 x 256
                const float* __restrict__ b_out,    // 16
                const _Float16* __restrict__ ws,
                float* __restrict__ out) {          // 16384 x 16
  __shared__ float s_phase[32];
  __shared__ float s_state[32][32];
  __shared__ float s_w[32][8];                       // dense per-cp blend wts
  __shared__ __align__(32) _Float16 s_x1[32][192];   // [exp|pn|state|pad0]
  __shared__ __align__(32) _Float16 s_h2in[32][288]; // [h1 -> h2 | state]

  const int tid   = threadIdx.x;
  const int lane  = tid & 31;
  const int wave  = tid >> 5;           // 8 waves
  const int row0  = blockIdx.x * 32;
  const int nsub  = lane & 15;          // N (or A-row) index within tile
  const int mbase = (lane >> 4) * 8;    // C/D row base (ISA 7.12.2)
  const int kbA   = (lane >> 4) * 16;   // A-fragment K base
  const int kbB   = (lane >> 4) * 16;   // B-fragment K base

  const _Float16* wsExp  = ws + OFF_EXPW;
  const _Float16* wsW1   = ws + OFF_W1;
  const _Float16* wsPf   = ws + OFF_PFW;
  const _Float16* wsWout = ws + OFF_WOUT;

  // ---- Phase 1: stage inputs (phase, state) into LDS ----
  for (int t = tid; t < 32 * 33; t += 256) {
    int r = t / 33, c = t % 33;
    float v = inputs[(row0 + r) * 33 + c];
    if (c == 0) {
      s_phase[r] = v;
    } else {
      s_state[r][c - 1] = v;
      s_x1[r][136 + (c - 1)]  = (_Float16)v;   // x1 state slot
      s_h2in[r][256 + (c - 1)] = (_Float16)v;  // h2_in state slot
    }
  }
  for (int t = tid; t < 32 * 24; t += 256)     // zero K pad 168..191
    s_x1[t / 24][168 + (t % 24)] = (_Float16)0.0f;
  __syncthreads();

  // ---- Phase 1b: cyclic Catmull-Rom -> dense wc[8] + phase_nodes ----
  if (tid < 32) {
    int r = tid;
    float f = s_phase[r] * 0.15915494309189535f;  // 1/(2*pi)
    f -= floorf(f);                               // remainder in [0,1)
    float pos = f * 8.0f;
    int base = (int)floorf(pos);
    if (base > 7) base = 7;
    float t = pos - (float)base;
    float t2 = t * t, t3 = t2 * t;
    float wk[4];
    wk[0] = -0.5f * t + t2 - 0.5f * t3;
    wk[1] = 1.0f - 2.5f * t2 + 1.5f * t3;
    wk[2] = 0.5f * t + 2.0f * t2 - 1.5f * t3;
    wk[3] = -0.5f * t2 + 0.5f * t3;
    float wc[8], pn[8];
#pragma unroll
    for (int c = 0; c < 8; ++c) wc[c] = 0.0f;
#pragma unroll
    for (int j = 0; j < 8; ++j) pn[j] = 0.0f;
#pragma unroll
    for (int k = 0; k < 4; ++k) {
      int ic = (base + 7 + k) & 7;                // (base-1+k) mod 8
      wc[ic] += wk[k];
#pragma unroll
      for (int j = 0; j < 8; ++j) pn[j] += wk[k] * cp[ic * 8 + j];
    }
#pragma unroll
    for (int c = 0; c < 8; ++c) s_w[r][c] = wc[c];
#pragma unroll
    for (int j = 0; j < 8; ++j) s_x1[r][128 + j] = (_Float16)pn[j];
  }
  __syncthreads();

  // ---- Phase 2a: expanded = elu(state @ exp_w.T + exp_b)  (K=32, N=128) ----
  {
    int n0 = wave * 16;  // each wave: one 16-wide N tile
    v16h b = *(const v16h*)(wsExp + (n0 + nsub) * 32 + kbB);
    float bb = exp_b[n0 + nsub];
#pragma unroll
    for (int mt = 0; mt < 2; ++mt) {
      v16h a = load_a_frag(&s_x1[mt * 16 + nsub][136 + kbA]);
      v8f acc = {};
      acc = wmma16(a, b, acc);
#pragma unroll
      for (int rr = 0; rr < 8; ++rr)
        s_x1[mt * 16 + mbase + rr][n0 + nsub] = (_Float16)eluf(acc[rr] + bb);
    }
  }
  __syncthreads();

  // ---- Phase 2b: h1 = elu(x1 @ w1.T + b1)  (K=192 padded, N=256) ----
#pragma unroll
  for (int nt = 0; nt < 2; ++nt) {
    int n0 = (wave * 2 + nt) * 16;
    float bb = b1[n0 + nsub];
#pragma unroll
    for (int mt = 0; mt < 2; ++mt) {
      v8f acc = {};
#pragma unroll
      for (int ks = 0; ks < 6; ++ks) {
        v16h a = load_a_frag(&s_x1[mt * 16 + nsub][ks * 32 + kbA]);
        v16h b = *(const v16h*)(wsW1 + (n0 + nsub) * 192 + ks * 32 + kbB);
        acc = wmma16(a, b, acc);
      }
#pragma unroll
      for (int rr = 0; rr < 8; ++rr)
        s_h2in[mt * 16 + mbase + rr][n0 + nsub] = (_Float16)eluf(acc[rr] + bb);
    }
  }
  __syncthreads();

  // ---- Phase 3: h2_pre = sum_c wc[c] * (h2_in @ pf_w[c].T + pf_b[c]) ----
  // (K=288, N=256, 8 control points; blend fused into epilogue)
  v8f fin[2][2] = {};
#pragma unroll
  for (int c = 0; c < 8; ++c) {
    v8f z[2][2] = {};
#pragma unroll
    for (int ks = 0; ks < 9; ++ks) {
      v16h a0 = load_a_frag(&s_h2in[nsub][ks * 32 + kbA]);
      v16h a1 = load_a_frag(&s_h2in[16 + nsub][ks * 32 + kbA]);
#pragma unroll
      for (int nt = 0; nt < 2; ++nt) {
        int n0 = (wave * 2 + nt) * 16;
        v16h b = *(const v16h*)(wsPf + (c * 256 + n0 + nsub) * 288 +
                                ks * 32 + kbB);
        z[0][nt] = wmma16(a0, b, z[0][nt]);
        z[1][nt] = wmma16(a1, b, z[1][nt]);
      }
    }
#pragma unroll
    for (int nt = 0; nt < 2; ++nt) {
      int n0 = (wave * 2 + nt) * 16;
      float pb = pf_b[c * 256 + n0 + nsub];
#pragma unroll
      for (int mt = 0; mt < 2; ++mt)
#pragma unroll
        for (int rr = 0; rr < 8; ++rr) {
          float wcv = s_w[mt * 16 + mbase + rr][c];
          fin[mt][nt][rr] += wcv * (z[mt][nt][rr] + pb);
        }
    }
  }
  __syncthreads();  // all reads of h1 region done before overwrite

  // h2 = elu(h2_pre); reuse s_h2in in place -> concat[h2, state]
#pragma unroll
  for (int nt = 0; nt < 2; ++nt) {
    int n0 = (wave * 2 + nt) * 16;
#pragma unroll
    for (int mt = 0; mt < 2; ++mt)
#pragma unroll
      for (int rr = 0; rr < 8; ++rr)
        s_h2in[mt * 16 + mbase + rr][n0 + nsub] =
            (_Float16)eluf(fin[mt][nt][rr]);
  }
  __syncthreads();

  // ---- Phase 4: out = concat[h2,state] @ w_out.T + b_out  (K=288, N=16) ----
  if (wave < 2) {  // wave-uniform branch: EXEC stays all-ones for WMMA
    int mt = wave;
    v8f acc = {};
#pragma unroll
    for (int ks = 0; ks < 9; ++ks) {
      v16h a = load_a_frag(&s_h2in[mt * 16 + nsub][ks * 32 + kbA]);
      v16h b = *(const v16h*)(wsWout + nsub * 288 + ks * 32 + kbB);
      acc = wmma16(a, b, acc);
    }
    float bo = b_out[nsub];
#pragma unroll
    for (int rr = 0; rr < 8; ++rr)
      out[(row0 + mt * 16 + mbase + rr) * 16 + nsub] = acc[rr] + bo;
  }
}

extern "C" void kernel_launch(void* const* d_in, const int* in_sizes, int n_in,
                              void* d_out, int out_size, void* d_ws,
                              size_t ws_size, hipStream_t stream) {
  (void)in_sizes; (void)n_in; (void)out_size; (void)ws_size;
  const float* inputs = (const float*)d_in[0];
  const float* exp_w  = (const float*)d_in[1];
  const float* exp_b  = (const float*)d_in[2];
  const float* cp     = (const float*)d_in[3];
  const float* w1     = (const float*)d_in[4];
  const float* b1     = (const float*)d_in[5];
  const float* pf_w   = (const float*)d_in[6];
  const float* pf_b   = (const float*)d_in[7];
  const float* w_out  = (const float*)d_in[8];
  const float* b_out  = (const float*)d_in[9];
  _Float16* ws = (_Float16*)d_ws;
  float* out = (float*)d_out;

  pfnn_convert_weights<<<256, 256, 0, stream>>>(exp_w, w1, pf_w, w_out, ws);
  pfnn_fused<<<16384 / 32, 256, 0, stream>>>(inputs, exp_b, cp, b1, pf_b,
                                             b_out, ws, out);
}
